// ProbabilisticAttention_21165598835368
// MI455X (gfx1250) — matrix operand order repaired and checked by reading, not compiled, not run on hardware
//
#include <hip/hip_runtime.h>
#include <hip/hip_bf16.h>

// ---------------------------------------------------------------------------
// ProbabilisticAttention for MI455X (gfx1250, wave32, WMMA).
//  Pass 1: six fused projections X@W+b via v_wmma_f32_16x16x32_bf16 with a
//          bf16-hi/lo split (3 WMMAs per K-step) for fp32-grade accuracy.
//          W tile staged in LDS column-major+padded so B fragments are
//          contiguous 32B runs -> ds_load_b128 pairs, zero bank conflicts.
//          A fragments are one contiguous 16-float run -> global_load_b128 x4.
//  Pass 2: per-position (b,s) fused Bhattacharyya distance + softmax + AV,
//          with the log-identity rewrite (1 v_log_f32 per distance element).
// ---------------------------------------------------------------------------

#define B_ 2
#define S_ 2048
#define D_ 1024
#define H_ 16
#define DKV_ 64
#define EPS_F 1e-6f
#define BETA_F 0.5f
#define SCALE_F 8.0f          // sqrt(64)
#define LOG4_F 1.3862943611198906f

#define COLPAD 40             // bf16 elems per LDS column (32 data + 8 pad)

typedef __attribute__((ext_vector_type(16))) __bf16 v16bf;
typedef __attribute__((ext_vector_type(8)))  __bf16 v8bf;
typedef __attribute__((ext_vector_type(8)))  float  v8f;
typedef __attribute__((ext_vector_type(4)))  float  v4f;

struct ProjArgs {
  const float* X[6];
  const float* W[6];
  const float* Bias[6];
  float*       O[6];
};

// ---------------------------------------------------------------------------
// Pass 1: O[z] = X[z] @ W[z] + Bias[z],  M=4096, N=1024, K=1024.
// Block = 256 threads = 8 waves. Block tile: 128(M) x 64(N).
// Wave tile: 16(M) x 64(N) -> 4 accumulators, K-step 32.
// bf16x3: acc += Ahi*Bhi + Ahi*Blo + Alo*Bhi  (fp32 accumulate).
// ---------------------------------------------------------------------------
__global__ __launch_bounds__(256) void proj_gemm_bf16x3(ProjArgs args) {
  const int z = blockIdx.z;
  const float* __restrict__ X    = args.X[z];
  const float* __restrict__ W    = args.W[z];
  const float* __restrict__ Bias = args.Bias[z];
  float*       __restrict__ O    = args.O[z];

  const int K = D_, N = D_;
  const int tid  = threadIdx.x;
  const int wave = tid >> 5;
  const int lane = tid & 31;
  const int half = lane >> 4;     // 0|1 : lane group within wave
  const int lid  = lane & 15;     // 0..15
  const int m0   = blockIdx.y * 128 + wave * 16;
  const int n0   = blockIdx.x * 64;

  // Staged W tile, column-major with pad: column c holds k=0..31 contiguous.
  // Column stride 40 bf16 = 80B -> fragment base banks tile 64 banks 4-apart.
  __shared__ __bf16 lbh[64 * COLPAD];
  __shared__ __bf16 lbl[64 * COLPAD];

  v8f acc[4] = {};   // 4 x 16x16 fp32 accumulators (wave covers 16x64)

  const int m  = m0 + lid;        // A row this lane owns
  const int sc = tid & 63;        // staging: column this thread fills
  const int sk = (tid >> 6) * 8;  // staging: k-run start (0/8/16/24)

  for (int kk = 0; kk < K; kk += 32) {
    __syncthreads();              // protect previous iteration's consumers

    // ---- stage W[kk:kk+32, n0:n0+64] -> LDS column-major, bf16 hi/lo.
    // Global reads: at fixed i, 64 consecutive threads read 256B rows of W.
    {
      float wv[8];
#pragma unroll
      for (int i = 0; i < 8; ++i) wv[i] = W[(kk + sk + i) * N + n0 + sc];
      v8bf ph, pl;
#pragma unroll
      for (int i = 0; i < 8; ++i) {
        __bf16 h = (__bf16)wv[i];
        ph[i] = h;
        pl[i] = (__bf16)(wv[i] - (float)h);
      }
      __builtin_memcpy(&lbh[sc * COLPAD + sk], &ph, 16);  // ds_store_b128
      __builtin_memcpy(&lbl[sc * COLPAD + sk], &pl, 16);  // ds_store_b128
    }

    // ---- A fragment: one contiguous 16-float run/lane:
    //      v=0..15 -> k = half*16 + v       (64B aligned)
    v16bf ahi, alo;
    {
      const float* ap = &X[m * K + kk + half * 16];
      v4f a0 = *(const v4f*)(ap + 0);
      v4f a1 = *(const v4f*)(ap + 4);
      v4f a2 = *(const v4f*)(ap + 8);
      v4f a3 = *(const v4f*)(ap + 12);
      float av[16];
#pragma unroll
      for (int i = 0; i < 4; ++i) {
        av[i]      = a0[i];
        av[i + 4]  = a1[i];
        av[i + 8]  = a2[i];
        av[i + 12] = a3[i];
      }
#pragma unroll
      for (int v = 0; v < 16; ++v) {
        __bf16 h = (__bf16)av[v];
        ahi[v] = h;
        alo[v] = (__bf16)(av[v] - (float)h);
      }
    }

    __syncthreads();              // W tile ready

    // ---- 4 N-tiles: contiguous 32B B fragments from LDS, 3 WMMAs each
#pragma unroll
    for (int t = 0; t < 4; ++t) {
      const int nc = t * 16 + lid;   // column within block tile
      v16bf bhi, blo;
      __builtin_memcpy(&bhi, &lbh[nc * COLPAD + half * 16], 32); // 2x ds_load_b128
      __builtin_memcpy(&blo, &lbl[nc * COLPAD + half * 16], 32); // 2x ds_load_b128
      acc[t] = __builtin_amdgcn_wmma_f32_16x16x32_bf16(
          false, ahi, false, bhi, (short)0, acc[t], false, false);
      acc[t] = __builtin_amdgcn_wmma_f32_16x16x32_bf16(
          false, ahi, false, blo, (short)0, acc[t], false, false);
      acc[t] = __builtin_amdgcn_wmma_f32_16x16x32_bf16(
          false, alo, false, bhi, (short)0, acc[t], false, false);
    }
  }

  // ---- epilogue: C/D layout (VGPR r -> M = r + 8*half, N = lid), add bias
#pragma unroll
  for (int t = 0; t < 4; ++t) {
    const int n = n0 + t * 16 + lid;
    const float bias = Bias[n];
#pragma unroll
    for (int r = 0; r < 8; ++r) {
      int mm = m0 + r + 8 * half;
      O[mm * N + n] = acc[t][r] + bias;
    }
  }
}

// ---------------------------------------------------------------------------
// Pass 2: one block (256 threads) per (b,s) position.
//  term1 rewrite: 0.25*log(0.25*(a/b+b/a+2)) = 0.25*(2log(a+b)-log a-log b-log4)
//  with a = sq^2+eps, b = sk^2+eps; per-row log-sums precomputed once.
//  term2 denom: sq^2+sk^2+eps = (a+b) - eps.
// ---------------------------------------------------------------------------
__global__ __launch_bounds__(256) void attn_fused(
    const float* __restrict__ mq, const float* __restrict__ sq,
    const float* __restrict__ mk, const float* __restrict__ sk,
    const float* __restrict__ mv, const float* __restrict__ sv,
    float* __restrict__ out_mu, float* __restrict__ out_sigma) {
  const int pos = blockIdx.x;          // 0..B*S-1
  const int b   = pos >> 11;           // S = 2048
  const int s   = pos & (S_ - 1);
  const int tid = threadIdx.x;

  __shared__ float s_mq[H_ * DKV_], s_mk[H_ * DKV_];
  __shared__ float s_aq[H_ * DKV_], s_ak[H_ * DKV_];   // sigma^2 + eps
  __shared__ float s_mv[H_ * DKV_], s_vv[H_ * DKV_];   // mu_v, sigma_v^2
  __shared__ float s_Lq[H_], s_Lk[H_];                 // row log-sums
  __shared__ float s_logit[H_][H_];
  __shared__ float s_attn[H_][H_];

  const int base = pos * D_;
#pragma unroll
  for (int rep = 0; rep < 4; ++rep) {
    int i = tid + rep * 256;           // 0..1023 == h*64+d
    float q  = sq[base + i];
    float k  = sk[base + i];
    float v  = sv[base + i];
    s_mq[i] = mq[base + i];
    s_mk[i] = mk[base + i];
    s_aq[i] = q * q + EPS_F;
    s_ak[i] = k * k + EPS_F;
    s_mv[i] = mv[base + i];
    s_vv[i] = v * v;
  }
  __syncthreads();

  // per-head log-sums: threads 0..15 -> q rows, 16..31 -> k rows
  if (tid < 32) {
    int h = tid & 15;
    const float* a = (tid < 16) ? &s_aq[h * DKV_] : &s_ak[h * DKV_];
    float acc = 0.f;
#pragma unroll 8
    for (int d = 0; d < DKV_; ++d) acc += logf(a[d]);
    if (tid < 16) s_Lq[h] = acc; else s_Lk[h] = acc;
  }
  __syncthreads();

  // distance for pair (i,j): one log + one divide per element
  const int hi = tid >> 4, hj = tid & 15;
  {
    const float* aq = &s_aq[hi * DKV_];
    const float* ak = &s_ak[hj * DKV_];
    const float* q  = &s_mq[hi * DKV_];
    const float* k  = &s_mk[hj * DKV_];
    float acc_log = 0.f, acc_t2 = 0.f;
#pragma unroll 4
    for (int d = 0; d < DKV_; ++d) {
      float sum2 = aq[d] + ak[d];              // sq^2 + sk^2 + 2eps
      acc_log += logf(sum2);
      float dm = q[d] - k[d];
      acc_t2  += dm * dm / (sum2 - EPS_F);     // denom = sq^2+sk^2+eps
    }
    float dist = 0.25f * (2.f * acc_log - s_Lq[hi] - s_Lk[hj]
                          - (float)DKV_ * LOG4_F)
               + 0.25f * acc_t2;
    s_logit[hi][hj] = -BETA_F * (dist / SCALE_F);
  }
  __syncthreads();

  // softmax over hj (16-wide rows)
  {
    float m = s_logit[hi][0];
#pragma unroll
    for (int j = 1; j < H_; ++j) m = fmaxf(m, s_logit[hi][j]);
    float sum = 0.f;
#pragma unroll
    for (int j = 0; j < H_; ++j) sum += expf(s_logit[hi][j] - m);
    s_attn[hi][hj] = expf(s_logit[hi][hj] - m) / sum;
  }
  __syncthreads();

  // out = attn @ mu_v / attn @ var_v ; store in [b][h][s][dk] order
#pragma unroll
  for (int rep = 0; rep < 4; ++rep) {
    int idx = tid + rep * 256;         // h*64 + d
    int h = idx >> 6, d = idx & 63;
    float am = 0.f, av = 0.f;
#pragma unroll
    for (int j = 0; j < H_; ++j) {
      float a = s_attn[h][j];
      am += a * s_mv[j * DKV_ + d];
      av += a * s_vv[j * DKV_ + d];
    }
    av = fmaxf(av, 0.f);
    int o = b * (S_ * D_) + h * (S_ * DKV_) + s * DKV_ + d;
    out_mu[o]    = am;
    out_sigma[o] = sqrtf(av + EPS_F);
  }
}

extern "C" void kernel_launch(void* const* d_in, const int* in_sizes, int n_in,
                              void* d_out, int out_size, void* d_ws, size_t ws_size,
                              hipStream_t stream) {
  (void)in_sizes; (void)n_in; (void)out_size; (void)ws_size;
  // Projection order in ws: 0 mu_q, 1 sigma_q, 2 mu_k, 3 sigma_k, 4 mu_v, 5 sigma_v
  ProjArgs args;
  const int xi[6] = {0, 3, 1, 4, 2, 5};      // query, query_sigma, key, key_sigma, value, value_sigma
  const int wi[6] = {6, 9, 7, 10, 8, 11};    // w_q_mu, w_q_sigma, w_k_mu, w_k_sigma, w_v_mu, w_v_sigma
  const int bi[6] = {12, 15, 13, 16, 14, 17};// b_q_mu, b_q_sigma, b_k_mu, b_k_sigma, b_v_mu, b_v_sigma
  float* ws = (float*)d_ws;
  const size_t PROJ = (size_t)B_ * S_ * D_;  // 4096*1024
  for (int i = 0; i < 6; ++i) {
    args.X[i]    = (const float*)d_in[xi[i]];
    args.W[i]    = (const float*)d_in[wi[i]];
    args.Bias[i] = (const float*)d_in[bi[i]];
    args.O[i]    = ws + (size_t)i * PROJ;
  }

  // Pass 1: grid (N/64, M/128, 6) = (16, 32, 6)
  proj_gemm_bf16x3<<<dim3(D_ / 64, (B_ * S_) / 128, 6), 256, 0, stream>>>(args);

  // Pass 2: one block per position
  float* out_mu    = (float*)d_out;
  float* out_sigma = out_mu + PROJ;
  attn_fused<<<B_ * S_, 256, 0, stream>>>(
      ws + 0 * PROJ, ws + 1 * PROJ, ws + 2 * PROJ,
      ws + 3 * PROJ, ws + 4 * PROJ, ws + 5 * PROJ,
      out_mu, out_sigma);
}
